// Gemma4TextExperts_87213605912610
// MI455X (gfx1250) — matrix operand order repaired and checked by reading, not compiled, not run on hardware
//
#include <hip/hip_runtime.h>
#include <hip/hip_bf16.h>

// Problem sizes (fixed by the reference)
#define TT 8192
#define HH 2048
#define II 4096
#define EE 8
#define KTOP 2
#define CAP (TT*KTOP)        // worst-case rows per expert = 16384
#define MTILE 64
#define MAXMT (CAP/MTILE)    // 256 worst-case M tiles per expert
#define NF 4                 // N-fragments per wave (64 cols)

typedef __attribute__((ext_vector_type(16))) __bf16 v16bf;
typedef __attribute__((ext_vector_type(8)))  __bf16 v8bf;
typedef __attribute__((ext_vector_type(8)))  float  v8f;

// ---------------------------------------------------------------- helpers ---

__device__ __forceinline__ v8f wmma_bf16(v16bf a, v16bf b, v8f c) {
  // D = A(16x32) * B(32x16) + C, f32 accumulate
  return __builtin_amdgcn_wmma_f32_16x16x32_bf16(
      /*neg_a=*/false, a, /*neg_b=*/false, b,
      /*c_mod=*/(short)0, c, /*reuse_a=*/false, /*reuse_b=*/false);
}

__device__ __forceinline__ v16bf pack16(float4 x0, float4 x1, float4 x2, float4 x3) {
  v16bf r;
  r[0]  = (__bf16)x0.x; r[1]  = (__bf16)x0.y; r[2]  = (__bf16)x0.z; r[3]  = (__bf16)x0.w;
  r[4]  = (__bf16)x1.x; r[5]  = (__bf16)x1.y; r[6]  = (__bf16)x1.z; r[7]  = (__bf16)x1.w;
  r[8]  = (__bf16)x2.x; r[9]  = (__bf16)x2.y; r[10] = (__bf16)x2.z; r[11] = (__bf16)x2.w;
  r[12] = (__bf16)x3.x; r[13] = (__bf16)x3.y; r[14] = (__bf16)x3.z; r[15] = (__bf16)x3.w;
  return r;
}

// A-fragment from f32 row: lane holds row M=(lane&15); 16 contiguous K
// {kb..kb+15} with kb = (lane>=16 ? 16 : 0)   (matches B contiguous order)
__device__ __forceinline__ v16bf loadA_f32(const float* __restrict__ rowp, int kk, int kbase) {
  const float4* q0 = (const float4*)(rowp + kk + kbase);
  const float4* q1 = (const float4*)(rowp + kk + kbase + 8);
  return pack16(q0[0], q0[1], q1[0], q1[1]);
}

// B-fragment from f32 weight row (gemm computes X @ W^T, so B(k,n) = W[n,k]):
// lane holds col N=(lane&15); 16 contiguous K at kk + (lane>=16 ? 16 : 0)
__device__ __forceinline__ v16bf loadB_f32(const float* __restrict__ p) {
  const float4* q = (const float4*)p;
  return pack16(q[0], q[1], q[2], q[3]);
}

// A-fragment from bf16 row (h buffer): two 16B runs per lane
__device__ __forceinline__ v16bf loadA_bf16(const __bf16* __restrict__ rowp, int kk, int kbase) {
  v8bf x0 = *(const v8bf*)(rowp + kk + kbase);
  v8bf x1 = *(const v8bf*)(rowp + kk + kbase + 8);
  return __builtin_shufflevector(x0, x1, 0,1,2,3,4,5,6,7,8,9,10,11,12,13,14,15);
}

// gelu_pytorch_tanh, branch-free:
//   0.5*x*(1+tanh(z)) == x * sigmoid(2z) == x / (1 + exp(-2z)),
//   z = sqrt(2/pi) * (x + 0.044715 x^3)
// Uses hardware v_exp_f32 / v_rcp_f32 (no divergent ocml slow path).
__device__ __forceinline__ float gelu_tanh_fast(float x) {
  const float c = 0.79788456080286535588f;          // sqrt(2/pi)
  const float n2log2e = -2.8853900817779268f;        // -2*log2(e)
  float z = c * (x + 0.044715f * x * x * x);
  float u = __builtin_amdgcn_exp2f(z * n2log2e);     // exp(-2z)
  return x * __builtin_amdgcn_rcpf(1.0f + u);
}

// -------------------------------------------------------------- routing -----
// One wave per expert: ordered ballot-compaction of (token, weight), pad to MTILE.
// meta layout: [0..E) count, [E..2E) padded, [2E..3E) compact row base
__global__ void moe_route(const int* __restrict__ idx, const float* __restrict__ tw,
                          int* __restrict__ meta, int* __restrict__ toks,
                          float* __restrict__ wts) {
  const int e = blockIdx.x;
  const int lane = threadIdx.x;              // wave32
  int cnt = 0;
  for (int t0 = 0; t0 < TT; t0 += 32) {
    const int t = t0 + lane;
#pragma unroll
    for (int k = 0; k < KTOP; ++k) {
      const bool m = (idx[t * KTOP + k] == e);
      unsigned long long mask = __ballot(m);
      int pos = cnt + __popcll(mask & ((1ull << lane) - 1ull));
      if (m) { toks[e * CAP + pos] = t; wts[e * CAP + pos] = tw[t * KTOP + k]; }
      cnt += __popcll(mask);
    }
  }
  const int padded = ((cnt + MTILE - 1) / MTILE) * MTILE;
  for (int p = cnt + lane; p < padded; p += 32) {   // pad rows: token 0, weight 0
    toks[e * CAP + p] = 0; wts[e * CAP + p] = 0.0f;
  }
  if (lane == 0) { meta[e] = cnt; meta[EE + e] = padded; }
}

__global__ void moe_prefix(int* __restrict__ meta) {
  int base = 0;
  for (int e = 0; e < EE; ++e) { meta[2 * EE + e] = base; base += meta[EE + e]; }
}

__global__ void moe_zero(float* __restrict__ out) {
  size_t i = (size_t)blockIdx.x * blockDim.x + threadIdx.x;
  if (i < (size_t)TT * HH) out[i] = 0.0f;
}

// ---------------------------------------------------------------- GEMM1 -----
// Per expert: h[row, i] = gelu_tanh(X@Wg^T) * (X@Wu^T), bf16 out into ws.
// Block = 64 rows x 128 i-cols; 4 waves (2M x 2N), each wave 32 rows x 64 cols
// (gate + up in parallel accumulators): 16 WMMA per K-step per wave.
__global__ __launch_bounds__(128)
void moe_gemm1(const float* __restrict__ X, const float* __restrict__ Wgu,
               const int* __restrict__ meta, const int* __restrict__ toks,
               __hip_bfloat16* __restrict__ hbuf) {
  const int e = blockIdx.z;
  const int padded = meta[EE + e];
  const int mt = blockIdx.y;
  if (mt * MTILE >= padded) return;          // uniform exit: EXEC stays full

  const int rowBase  = mt * MTILE;
  const int hRowBase = meta[2 * EE + e] + rowBase;
  const int i0 = blockIdx.x * 128;
  const float* __restrict__ W = Wgu + (size_t)e * (2 * II) * HH;   // [2I, H]

  const int wave = threadIdx.x >> 5, lane = threadIdx.x & 31;
  const int wm = (wave >> 1) * 32, wn = (wave & 1) * 64;
  const int ln = lane & 15, hi = lane >> 4;
  const int kbaseA = hi * 16;                // A layout K split
  const int kbaseB = hi * 16;                // B layout K split

  const int tok0 = toks[e * CAP + rowBase + wm + ln];
  const int tok1 = toks[e * CAP + rowBase + wm + 16 + ln];
  const float* __restrict__ xp0 = X + (size_t)tok0 * HH;
  const float* __restrict__ xp1 = X + (size_t)tok1 * HH;

  const float* bg[NF];
  const float* bu[NF];
#pragma unroll
  for (int f = 0; f < NF; ++f) {
    const int ng = i0 + wn + 16 * f + ln;    // gate row of W; up row = ng + I
    bg[f] = W + (size_t)ng * HH;
    bu[f] = W + (size_t)(ng + II) * HH;
  }

  v8f accG[2][NF] = {};
  v8f accU[2][NF] = {};

  for (int kk = 0; kk < HH; kk += 32) {
    __builtin_prefetch(bg[0] + kk + 256, 0, 1);
    __builtin_prefetch(bu[0] + kk + 256, 0, 1);
    v16bf a0 = loadA_f32(xp0, kk, kbaseA);
    v16bf a1 = loadA_f32(xp1, kk, kbaseA);
#pragma unroll
    for (int f = 0; f < NF; ++f) {
      v16bf g = loadB_f32(bg[f] + kk + kbaseB);
      accG[0][f] = wmma_bf16(a0, g, accG[0][f]);
      accG[1][f] = wmma_bf16(a1, g, accG[1][f]);
      v16bf u = loadB_f32(bu[f] + kk + kbaseB);
      accU[0][f] = wmma_bf16(a0, u, accU[0][f]);
      accU[1][f] = wmma_bf16(a1, u, accU[1][f]);
    }
  }

  // epilogue: h = gelu(gate) * up, store bf16 (C layout: M = 16*fm + r + 8*hi, N = ln)
#pragma unroll
  for (int fm = 0; fm < 2; ++fm)
#pragma unroll
    for (int f = 0; f < NF; ++f)
#pragma unroll
      for (int r = 0; r < 8; ++r) {
        float g = accG[fm][f][r], u = accU[fm][f][r];
        float hv = gelu_tanh_fast(g) * u;
        int mrow = wm + 16 * fm + r + 8 * hi;
        size_t off = (size_t)(hRowBase + mrow) * II + (i0 + wn + 16 * f + ln);
        hbuf[off] = __float2bfloat16(hv);
      }
}

// ---------------------------------------------------------------- GEMM2 -----
// out[tok, n] += w(row) * sum_i h[row, i] * down[e, n, i]; atomic scatter.
// Block = 64 rows x 128 out-cols; 4 waves (2M x 2N), wave = 32x64: 8 WMMA/K-step.
__global__ __launch_bounds__(128)
void moe_gemm2(const __hip_bfloat16* __restrict__ hbuf, const float* __restrict__ Wd,
               const int* __restrict__ meta, const int* __restrict__ toks,
               const float* __restrict__ wts, float* __restrict__ out) {
  const int e = blockIdx.z;
  const int padded = meta[EE + e];
  const int mt = blockIdx.y;
  if (mt * MTILE >= padded) return;

  const int rowBase  = mt * MTILE;
  const int hRowBase = meta[2 * EE + e] + rowBase;
  const int n0 = blockIdx.x * 128;
  const float* __restrict__ Wde = Wd + (size_t)e * HH * II;        // [H, I]

  const int wave = threadIdx.x >> 5, lane = threadIdx.x & 31;
  const int wm = (wave >> 1) * 32, wn = (wave & 1) * 64;
  const int ln = lane & 15, hi = lane >> 4;
  const int kbaseA = hi * 16;
  const int kbaseB = hi * 16;

  const __bf16* __restrict__ hb = reinterpret_cast<const __bf16*>(hbuf);
  const __bf16* __restrict__ hp0 = hb + (size_t)(hRowBase + wm + ln) * II;
  const __bf16* __restrict__ hp1 = hb + (size_t)(hRowBase + wm + 16 + ln) * II;

  const float* bd[NF];
#pragma unroll
  for (int f = 0; f < NF; ++f) {
    const int nr = n0 + wn + 16 * f + ln;
    bd[f] = Wde + (size_t)nr * II;
  }

  v8f acc[2][NF] = {};

  for (int kk = 0; kk < II; kk += 32) {
    __builtin_prefetch(bd[0] + kk + 256, 0, 1);
    __builtin_prefetch(bd[2] + kk + 256, 0, 1);
    v16bf a0 = loadA_bf16(hp0, kk, kbaseA);
    v16bf a1 = loadA_bf16(hp1, kk, kbaseA);
#pragma unroll
    for (int f = 0; f < NF; ++f) {
      v16bf b = loadB_f32(bd[f] + kk + kbaseB);
      acc[0][f] = wmma_bf16(a0, b, acc[0][f]);
      acc[1][f] = wmma_bf16(a1, b, acc[1][f]);
    }
  }

  // scatter-add with routing weight (pad rows have weight 0 -> skipped)
#pragma unroll
  for (int fm = 0; fm < 2; ++fm)
#pragma unroll
    for (int r = 0; r < 8; ++r) {
      int mrow = wm + 16 * fm + r + 8 * hi;
      int li = rowBase + mrow;
      float wgt = wts[e * CAP + li];
      if (wgt != 0.0f) {
        int tokid = toks[e * CAP + li];
        float* op = out + (size_t)tokid * HH + (n0 + wn);
#pragma unroll
        for (int f = 0; f < NF; ++f)
          atomicAdd(op + 16 * f + ln, acc[fm][f][r] * wgt);
      }
    }
}

// ------------------------------------------------------------------ host ----
extern "C" void kernel_launch(void* const* d_in, const int* in_sizes, int n_in,
                              void* d_out, int out_size, void* d_ws, size_t ws_size,
                              hipStream_t stream) {
  const float* X   = (const float*)d_in[0];   // [T, H]
  const int*   idx = (const int*)  d_in[1];   // [T, K]
  const float* tw  = (const float*)d_in[2];   // [T, K]
  const float* Wgu = (const float*)d_in[3];   // [E, 2I, H]
  const float* Wd  = (const float*)d_in[4];   // [E, H, I]
  float* out = (float*)d_out;                 // [T, H]

  // workspace layout: meta (256B) | token lists | weight lists | h (bf16)
  char* ws = (char*)d_ws;
  int*   meta = (int*)ws;
  int*   toks = (int*)(ws + 256);
  float* wlst = (float*)(ws + 256 + (size_t)EE * CAP * 4);
  __hip_bfloat16* hbuf =
      (__hip_bfloat16*)(ws + 256 + 2 * (size_t)EE * CAP * 4);
  // h worst case: (T*K + E*MTILE) * I * 2B ~= 138 MB

  moe_route <<<EE, 32, 0, stream>>>(idx, tw, meta, toks, wlst);
  moe_prefix<<<1, 1, 0, stream>>>(meta);

  const size_t nOut = (size_t)TT * HH;
  moe_zero<<<(unsigned)((nOut + 255) / 256), 256, 0, stream>>>(out);

  dim3 g1(II / 128, MAXMT, EE);
  moe_gemm1<<<g1, 128, 0, stream>>>(X, Wgu, meta, toks, hbuf);

  dim3 g2(HH / 128, MAXMT, EE);
  moe_gemm2<<<g2, 128, 0, stream>>>(hbuf, Wd, meta, toks, wlst, out);
}
